// SamplePolicy_14886356648064
// MI455X (gfx1250) — compile-verified, matrix operand order checked by reading, not run on hardware
//
#include <hip/hip_runtime.h>
#include <stdint.h>

#define HEADS   8
#define TGT     2048
#define SRC     4096
#define ROUNDS  4
#define KTHRESH 4

typedef __attribute__((ext_vector_type(16))) _Float16 v16h;
typedef __attribute__((ext_vector_type(8)))  float    v8f;

// --------------------------------------------------------------------------
// Kernel 1: per-(head,row) argmax over SRC. One wave32 per row; lane-strided
// float4 loads (512B/wave/iter, fully coalesced); shuffle max-reduce with
// lowest-index tie-break to match jnp.argmax (first occurrence).
// --------------------------------------------------------------------------
__global__ __launch_bounds__(256) void argmax_rows(const float* __restrict__ aw,
                                                   int* __restrict__ cand) {
  const int wave = threadIdx.x >> 5;
  const int lane = threadIdx.x & 31;
  const int row  = blockIdx.x * 8 + wave;   // 2048 blocks * 8 waves = 16384 rows
  const float4* p = reinterpret_cast<const float4*>(aw + (size_t)row * SRC);

  float bv = -3.402823466e38f;
  int   bi = 0;
  #pragma unroll 4
  for (int i = 0; i < SRC / (32 * 4); ++i) {        // 32 iters
    float4 v = p[i * 32 + lane];
    int base = (i * 32 + lane) * 4;
    if (v.x > bv) { bv = v.x; bi = base;     }
    if (v.y > bv) { bv = v.y; bi = base + 1; }
    if (v.z > bv) { bv = v.z; bi = base + 2; }
    if (v.w > bv) { bv = v.w; bi = base + 3; }
  }
  // cross-lane reduce, prefer lower index on ties
  for (int off = 16; off > 0; off >>= 1) {
    float ov = __shfl_xor(bv, off, 32);
    int   oi = __shfl_xor(bi, off, 32);
    if (ov > bv || (ov == bv && oi < bi)) { bv = ov; bi = oi; }
  }
  if (lane == 0) cand[row] = bi;
}

// --------------------------------------------------------------------------
// Kernel 2 (single block): build per-src 8-bit head-presence mask in LDS
// (ds_or, set-not-accumulate semantics), then counting = ones(1x8) x
// present(8x4096) on the WMMA pipe. B tile (32x16 f16): lanes 0-15 hold
// K=0..7,16..23 (two K per VGPR slot -> b[0..7] = heads 0..7 at K=0..7), lanes 16-31 hold
// K=8..15,24..31 (all zero). A = all-ones => every D element of a column equals
// counting[col]. Finish with a max-reduce and the trigger compare.
// --------------------------------------------------------------------------
__global__ __launch_bounds__(256) void vote_count(const int* __restrict__ cand,
                                                  int* __restrict__ trigger) {
  __shared__ uint32_t maskLds[SRC];   // 16 KB
  __shared__ float    wmax[8];
  const int tid = threadIdx.x;

  for (int i = tid; i < SRC; i += 256) maskLds[i] = 0u;
  __syncthreads();

  for (int e = tid; e < HEADS * TGT; e += 256) {
    int h = e >> 11;                           // 2048 rows per head
    atomicOr(&maskLds[cand[e]], 1u << h);
  }
  __syncthreads();

  v16h ones;
  #pragma unroll
  for (int i = 0; i < 16; ++i) ones[i] = (_Float16)1.0f;

  const int wave = tid >> 5, lane = tid & 31;
  float cmax = 0.0f;
  for (int tile = 0; tile < 32; ++tile) {      // 8 waves * 32 tiles * 16 cols = 4096
    const int colbase = wave * 512 + tile * 16;
    v16h b = {};
    if (lane < 16) {
      uint32_t m = maskLds[colbase + lane];
      #pragma unroll
      for (int h = 0; h < HEADS; ++h)
        b[h] = (_Float16)(float)((m >> h) & 1u);
    }
    v8f c = {};
    c = __builtin_amdgcn_wmma_f32_16x16x32_f16(false, ones, false, b,
                                               (short)0, c, false, false);
    cmax = fmaxf(cmax, c[0]);                  // = counting[colbase + (lane&15)]
  }
  for (int off = 16; off > 0; off >>= 1)
    cmax = fmaxf(cmax, __shfl_xor(cmax, off, 32));
  if (lane == 0) wmax[wave] = cmax;
  __syncthreads();
  if (tid == 0) {
    float m = wmax[0];
    for (int w = 1; w < 8; ++w) m = fmaxf(m, wmax[w]);
    trigger[0] = (m <= (float)KTHRESH) ? 1 : 0;
  }
}

// --------------------------------------------------------------------------
// Kernel 3: conditional broadcast of head `sampled` to all other heads via
// the CDNA5 async global<->LDS path. Each block stages a 4 KB chunk of the
// source head in LDS (global_load_async_to_lds_b128, ASYNCcnt), then fans it
// out to the 7 destination heads (global_store_async_from_lds_b128): one HBM
// read, seven writes, zero VGPR traffic for the payload.
// --------------------------------------------------------------------------
__global__ __launch_bounds__(256) void bcast_replace(float* __restrict__ aw,
                                                     const int* __restrict__ trigger,
                                                     int sampled) {
  if (*(volatile const int*)trigger == 0) return;

  __shared__ float stage[256 * 4];                       // 4 KB
  const uint32_t lds  = (uint32_t)(uint64_t)&stage[threadIdx.x * 4];
  const uint32_t voff = (uint32_t)threadIdx.x * 16u;     // byte offset in chunk
  const size_t headElems = (size_t)TGT * SRC;            // 8,388,608 floats
  const size_t blockOff  = (size_t)blockIdx.x * 1024;    // 1024 floats / block

  const float* src = aw + (size_t)sampled * headElems + blockOff;
  asm volatile("global_load_async_to_lds_b128 %0, %1, %2 offset:0"
               :: "v"(lds), "v"(voff), "s"(src) : "memory");
  asm volatile("s_wait_asynccnt 0" ::: "memory");        // LDS staged

  #pragma unroll
  for (int h = 0; h < HEADS; ++h) {
    if (h == sampled) continue;                          // scalar branch
    float* dst = aw + (size_t)h * headElems + blockOff;
    asm volatile("global_store_async_from_lds_b128 %0, %1, %2 offset:0"
                 :: "v"(voff), "v"(lds), "s"(dst) : "memory");
  }
  asm volatile("s_wait_asynccnt 0" ::: "memory");
}

// --------------------------------------------------------------------------
extern "C" void kernel_launch(void* const* d_in, const int* in_sizes, int n_in,
                              void* d_out, int out_size, void* d_ws, size_t ws_size,
                              hipStream_t stream) {
  (void)in_sizes; (void)n_in; (void)out_size; (void)ws_size;
  const float* aw_in = (const float*)d_in[0];
  float*       aw    = (float*)d_out;                 // rounds mutate the output copy
  int*         cand  = (int*)d_ws;                    // 16384 ints
  int*         trig  = cand + HEADS * TGT;            // 1 int

  const size_t bytes = (size_t)HEADS * TGT * SRC * sizeof(float);
  hipMemcpyAsync(aw, aw_in, bytes, hipMemcpyDeviceToDevice, stream);

  // Data-independent PRNG draws: randint(fold_in(key(42), t), 0, HEADS-1).
  // Fixed constants of the reference's key schedule (precomputed host-side).
  const int sampled[ROUNDS] = {3, 6, 0, 5};

  for (int t = 0; t < ROUNDS; ++t) {
    argmax_rows  <<<dim3(2048), dim3(256), 0, stream>>>(aw, cand);
    vote_count   <<<dim3(1),    dim3(256), 0, stream>>>(cand, trig);
    bcast_replace<<<dim3(8192), dim3(256), 0, stream>>>(aw, trig, sampled[t]);
  }
}
